// LSTMModel_57836029608428
// MI455X (gfx1250) — matrix operand order repaired and checked by reading, not compiled, not run on hardware
//
#include <hip/hip_runtime.h>
#include <math.h>

// ---------------------------------------------------------------------------
// Types for CDNA5 WMMA (wave32)
// ---------------------------------------------------------------------------
typedef __bf16 bf16;
typedef __attribute__((ext_vector_type(16))) __bf16 v16bf;
typedef __attribute__((ext_vector_type(8)))  float  v8f;

static constexpr int Bsz = 16;
static constexpr int T   = 128;
static constexpr int D   = 1024;
static constexpr int G   = 4 * D;       // 4096 gate width
static constexpr int V   = 32000;
static constexpr int M   = T * Bsz;     // 2048 rows of X / H / out

// round-to-nearest-even f32 -> bf16 (bit-exact, no reliance on cast support)
__device__ inline bf16 to_bf16(float f) {
    union { float f; unsigned u; } in; in.f = f;
    unsigned u = in.u;
    unsigned r = (u + 0x7FFFu + ((u >> 16) & 1u)) >> 16;
    union { unsigned short s; bf16 b; } out; out.s = (unsigned short)r;
    return out.b;
}

// ---------------------------------------------------------------------------
// WMMA fragment loaders (ISA 7.12.2 layouts, K-contiguous memory)
// A: [M,K] row-major, 16x32 tile.  lanes 0-15: K0..7 (v0-3), K16..23 (v4-7)
//                                  lanes16-31: K8..15, K24..31
// B: stored [N,K] row-major (math B^T), 32x16 tile. lane n holds col n:
//    lanes 0-15: K0..7, K16..23, lanes 16-31: K8..15, K24..31
// ---------------------------------------------------------------------------
__device__ inline v16bf load_a_frag(const bf16* __restrict__ A, int ldk,
                                    int row_base, int k0, int lane) {
    union { uint4 u[2]; v16bf v; } f;
    int r = row_base + (lane & 15);
    const bf16* p = A + (long)r * ldk + k0 + ((lane >> 4) << 3);
    f.u[0] = *reinterpret_cast<const uint4*>(p);
    f.u[1] = *reinterpret_cast<const uint4*>(p + 16);
    return f.v;
}

__device__ inline v16bf load_b_frag(const bf16* __restrict__ B, int ldk,
                                    int col_base, int k0, int lane) {
    union { uint4 u[2]; v16bf v; } f;
    int c = col_base + (lane & 15);
    const bf16* p = B + (long)c * ldk + k0 + ((lane >> 4) << 3);
    f.u[0] = *reinterpret_cast<const uint4*>(p);
    f.u[1] = *reinterpret_cast<const uint4*>(p + 16);
    return f.v;
}

// ---------------------------------------------------------------------------
// Elementwise fp32 -> bf16 cast
// ---------------------------------------------------------------------------
__global__ __launch_bounds__(256)
void cast_f32_bf16(const float* __restrict__ src, bf16* __restrict__ dst, long n) {
    long i = (long)blockIdx.x * blockDim.x + threadIdx.x;
    long stride = (long)gridDim.x * blockDim.x;
    for (; i < n; i += stride) dst[i] = to_bf16(src[i]);
}

// ---------------------------------------------------------------------------
// Embedding gather + cast: X[t*16+b, :] = bf16(emb[idx[b,t], :])
// ---------------------------------------------------------------------------
__global__ __launch_bounds__(256)
void gather_cast(const float* __restrict__ emb, const int* __restrict__ idx,
                 bf16* __restrict__ X) {
    int m = blockIdx.x;            // m = t*16 + b
    int t = m >> 4, b = m & 15;
    int tok = idx[b * T + t];
    const float* src = emb + (long)tok * D;
    bf16* dst = X + (long)m * D;
    for (int d = threadIdx.x; d < D; d += 256) dst[d] = to_bf16(src[d]);
}

// ---------------------------------------------------------------------------
// Generic bf16 GEMM:  C[M,N] = A[M,K] @ B[N,K]^T + bias[N]   (f32 accumulate)
// Block = 8 waves; wave computes a 16x128 tile (8x v_wmma_f32_16x16x32_bf16
// per K-step). Software-pipelined: iteration k issues the loads for k+1
// before consuming iteration k's fragments, so WMMAs overlap memory.
// gridDim = (ceil(N/1024), M/16).
// ---------------------------------------------------------------------------
__global__ __launch_bounds__(256)
void gemm_bf16_nt(const bf16* __restrict__ A, const bf16* __restrict__ B,
                  const float* __restrict__ bias, float* __restrict__ C,
                  int Mm, int Nn, int Kk) {
    const int wave = threadIdx.x >> 5;
    const int lane = threadIdx.x & 31;
    const int m0 = blockIdx.y * 16;
    const int n0 = (blockIdx.x * 8 + wave) * 128;
    if (n0 >= Nn || m0 >= Mm) return;

    v8f acc[8];
    #pragma unroll
    for (int tt = 0; tt < 8; ++tt) {
        float bv = bias[n0 + tt * 16 + (lane & 15)];
        #pragma unroll
        for (int r = 0; r < 8; ++r) acc[tt][r] = bv;
    }

    // prologue: fragments for k0 = 0
    v16bf af = load_a_frag(A, Kk, m0, 0, lane);
    v16bf bfr[8];
    #pragma unroll
    for (int tt = 0; tt < 8; ++tt)
        bfr[tt] = load_b_frag(B, Kk, n0 + tt * 16, 0, lane);

    // pipelined main loop (peeled epilogue)
    for (int k0 = 0; k0 < Kk - 32; k0 += 32) {
        const int kn = k0 + 32;
        v16bf afn = load_a_frag(A, Kk, m0, kn, lane);
        v16bf bfn[8];
        #pragma unroll
        for (int tt = 0; tt < 8; ++tt)
            bfn[tt] = load_b_frag(B, Kk, n0 + tt * 16, kn, lane);

        #pragma unroll
        for (int tt = 0; tt < 8; ++tt)
            acc[tt] = __builtin_amdgcn_wmma_f32_16x16x32_bf16(
                false, af, false, bfr[tt], (short)0, acc[tt], false, false);

        af = afn;
        #pragma unroll
        for (int tt = 0; tt < 8; ++tt) bfr[tt] = bfn[tt];
    }
    #pragma unroll
    for (int tt = 0; tt < 8; ++tt)
        acc[tt] = __builtin_amdgcn_wmma_f32_16x16x32_bf16(
            false, af, false, bfr[tt], (short)0, acc[tt], false, false);

    const int rowBase = m0 + ((lane >> 4) << 3);
    #pragma unroll
    for (int tt = 0; tt < 8; ++tt) {
        long cb = (long)(n0 + tt * 16 + (lane & 15));
        #pragma unroll
        for (int r = 0; r < 8; ++r)
            C[(long)(rowBase + r) * Nn + cb] = acc[tt][r];
    }
}

// ---------------------------------------------------------------------------
// Init double-buffered h state (zeros) + the inter-block barrier counter
// ---------------------------------------------------------------------------
__global__ __launch_bounds__(256)
void init_state(bf16* __restrict__ hbuf, unsigned* __restrict__ bar) {
    long i = (long)blockIdx.x * blockDim.x + threadIdx.x;
    if (i == 0) *bar = 0u;
    if (i < 2L * Bsz * D) hbuf[i] = to_bf16(0.0f);
}

// ---------------------------------------------------------------------------
// Sequential LSTM recurrence. 16 blocks; block j owns h columns [64j, 64j+64).
// Per step: gates[16,256] = Xg slice + h @ Wh^T slice (WMMA), cell update,
// bf16 h written to the alternate global buffer + H_all, then a global
// atomic barrier synchronizes all 16 blocks.
// ---------------------------------------------------------------------------
__global__ __launch_bounds__(256)
void lstm_recur(const float* __restrict__ Xg, const bf16* __restrict__ Whb,
                bf16* __restrict__ hbuf, bf16* __restrict__ Hall,
                unsigned* __restrict__ bar) {
    __shared__ float gates[16][257];   // 4 chunks x 64 local cols (+pad)
    __shared__ float cst[16][64];      // persistent cell state slice

    const int tid  = threadIdx.x;
    const int lane = tid & 31;
    const int wave = tid >> 5;
    const int jBase = blockIdx.x * 64;

    for (int q = tid; q < 16 * 64; q += 256) cst[q >> 6][q & 63] = 0.0f;
    __syncthreads();

    const int tt0 = wave * 2, tt1 = wave * 2 + 1;
    const int nc0 = (tt0 >> 2) * D + jBase + (tt0 & 3) * 16;
    const int nc1 = (tt1 >> 2) * D + jBase + (tt1 & 3) * 16;

    for (int t = 0; t < T; ++t) {
        const bf16* hrd = hbuf + (long)(t & 1) * Bsz * D;
        bf16*       hwr = hbuf + (long)((t + 1) & 1) * Bsz * D;

        // init accumulators with precomputed x-projection (+Wx_b)
        v8f acc0, acc1;
        {
            const int rb = (lane >> 4) << 3;
            long base0 = ((long)(t * 16 + rb)) * G + nc0 + (lane & 15);
            long base1 = ((long)(t * 16 + rb)) * G + nc1 + (lane & 15);
            #pragma unroll
            for (int r = 0; r < 8; ++r) {
                acc0[r] = Xg[base0 + (long)r * G];
                acc1[r] = Xg[base1 + (long)r * G];
            }
        }

        // software-pipelined K loop over h @ Wh^T
        v16bf af = load_a_frag(hrd, D, 0, 0, lane);
        v16bf b0 = load_b_frag(Whb, D, nc0, 0, lane);
        v16bf b1 = load_b_frag(Whb, D, nc1, 0, lane);
        for (int k0 = 0; k0 < D - 32; k0 += 32) {
            v16bf afn = load_a_frag(hrd, D, 0, k0 + 32, lane);
            v16bf b0n = load_b_frag(Whb, D, nc0, k0 + 32, lane);
            v16bf b1n = load_b_frag(Whb, D, nc1, k0 + 32, lane);
            acc0 = __builtin_amdgcn_wmma_f32_16x16x32_bf16(
                false, af, false, b0, (short)0, acc0, false, false);
            acc1 = __builtin_amdgcn_wmma_f32_16x16x32_bf16(
                false, af, false, b1, (short)0, acc1, false, false);
            af = afn; b0 = b0n; b1 = b1n;
        }
        acc0 = __builtin_amdgcn_wmma_f32_16x16x32_bf16(
            false, af, false, b0, (short)0, acc0, false, false);
        acc1 = __builtin_amdgcn_wmma_f32_16x16x32_bf16(
            false, af, false, b1, (short)0, acc1, false, false);

        {
            const int rb = (lane >> 4) << 3;
            const int c0 = tt0 * 16 + (lane & 15);
            const int c1 = tt1 * 16 + (lane & 15);
            #pragma unroll
            for (int r = 0; r < 8; ++r) {
                gates[rb + r][c0] = acc0[r];
                gates[rb + r][c1] = acc1[r];
            }
        }
        __syncthreads();

        // elementwise cell update for 16x64 slice
        for (int q = tid; q < 16 * 64; q += 256) {
            int b = q >> 6, jl = q & 63;
            float iv = gates[b][jl];
            float fv = gates[b][64 + jl];
            float ov = gates[b][128 + jl];
            float gv = gates[b][192 + jl];
            iv = 1.0f / (1.0f + __expf(-iv));
            fv = 1.0f / (1.0f + __expf(-fv));
            ov = 1.0f / (1.0f + __expf(-ov));
            gv = tanhf(gv);
            float cn = fv * cst[b][jl] + iv * gv;
            cst[b][jl] = cn;
            float hn = ov * tanhf(cn);
            bf16 hb = to_bf16(hn);
            hwr[b * D + jBase + jl] = hb;
            Hall[((long)(t * 16 + b)) * D + jBase + jl] = hb;
        }

        // global barrier across the 16 blocks (monotonic counter)
        __threadfence();
        __syncthreads();
        if (tid == 0) {
            atomicAdd(bar, 1u);
            while (atomicAdd(bar, 0u) < 16u * (unsigned)(t + 1)) {
                __builtin_amdgcn_s_sleep(1);
            }
        }
        __syncthreads();
    }
}

// ---------------------------------------------------------------------------
// In-place row-wise log-softmax over V=32000 (one block per row)
// ---------------------------------------------------------------------------
__global__ __launch_bounds__(256)
void log_softmax_rows(float* __restrict__ out) {
    __shared__ float red[256];
    float* row = out + (long)blockIdx.x * V;
    const int tid = threadIdx.x;

    float m = -INFINITY;
    for (int v = tid; v < V; v += 256) m = fmaxf(m, row[v]);
    red[tid] = m; __syncthreads();
    for (int s = 128; s > 0; s >>= 1) {
        if (tid < s) red[tid] = fmaxf(red[tid], red[tid + s]);
        __syncthreads();
    }
    float mx = red[0]; __syncthreads();

    float sum = 0.0f;
    for (int v = tid; v < V; v += 256) sum += __expf(row[v] - mx);
    red[tid] = sum; __syncthreads();
    for (int s = 128; s > 0; s >>= 1) {
        if (tid < s) red[tid] += red[tid + s];
        __syncthreads();
    }
    float lse = mx + __logf(red[0]);

    for (int v = tid; v < V; v += 256) row[v] -= lse;
}

// ---------------------------------------------------------------------------
// Host-side orchestration
// ---------------------------------------------------------------------------
extern "C" void kernel_launch(void* const* d_in, const int* in_sizes, int n_in,
                              void* d_out, int out_size, void* d_ws, size_t ws_size,
                              hipStream_t stream) {
    const int*   idx  = (const int*)  d_in[0];
    const float* emb  = (const float*)d_in[1];
    const float* Wx_w = (const float*)d_in[2];
    const float* Wx_b = (const float*)d_in[3];
    const float* Wh_w = (const float*)d_in[4];
    const float* fc_w = (const float*)d_in[5];
    const float* fc_b = (const float*)d_in[6];
    float* out = (float*)d_out;

    char* ws = (char*)d_ws;
    size_t off = 0;
    auto alloc = [&](size_t bytes) -> void* {
        void* p = ws + off;
        off += (bytes + 255) & ~(size_t)255;
        return p;
    };

    bf16*     Wxb  = (bf16*)alloc((size_t)G * D * sizeof(bf16));        // 8 MB
    bf16*     Whb  = (bf16*)alloc((size_t)G * D * sizeof(bf16));        // 8 MB
    bf16*     fcb  = (bf16*)alloc((size_t)V * D * sizeof(bf16));        // 64 MB
    bf16*     X    = (bf16*)alloc((size_t)M * D * sizeof(bf16));        // 4 MB
    float*    Xg   = (float*)alloc((size_t)M * G * sizeof(float));      // 32 MB
    bf16*     Hall = (bf16*)alloc((size_t)M * D * sizeof(bf16));        // 4 MB
    bf16*     hbuf = (bf16*)alloc((size_t)2 * Bsz * D * sizeof(bf16));  // 64 KB
    unsigned* bar  = (unsigned*)alloc(256);

    // 1) weight casts to bf16
    cast_f32_bf16<<<2048, 256, 0, stream>>>(Wx_w, Wxb, (long)G * D);
    cast_f32_bf16<<<2048, 256, 0, stream>>>(Wh_w, Whb, (long)G * D);
    cast_f32_bf16<<<4096, 256, 0, stream>>>(fc_w, fcb, (long)V * D);

    // 2) embedding gather (+ time-major reorder) to bf16
    gather_cast<<<M, 256, 0, stream>>>(emb, idx, X);

    // 3) batched input projection: Xg = X @ Wx^T + Wx_b  [2048 x 4096]
    gemm_bf16_nt<<<dim3(G / 1024, M / 16), 256, 0, stream>>>(X, Wxb, Wx_b, Xg, M, G, D);

    // 4) recurrence state init + sequential LSTM steps
    init_state<<<(2 * Bsz * D + 255) / 256, 256, 0, stream>>>(hbuf, bar);
    lstm_recur<<<16, 256, 0, stream>>>(Xg, Whb, hbuf, Hall, bar);

    // 5) classifier GEMM: out = H @ fc^T + fc_b  [2048 x 32000]
    gemm_bf16_nt<<<dim3((V + 1023) / 1024, M / 16), 256, 0, stream>>>(Hall, fcb, fc_b, out, M, V, D);

    // 6) in-place row-wise log-softmax
    log_softmax_rows<<<M, 256, 0, stream>>>(out);
}
